// Attention_36386962931993
// MI455X (gfx1250) — compile-verified, matrix operand order checked by reading, not run on hardware
//
#include <hip/hip_runtime.h>
#include <hip/hip_bf16.h>

// Attention with softmax over the QUERY axis (axis 0).
//   colmax[m] = max_n (Q[n]·K[m]),  colsum[m] = sum_n exp(s - colmax[m])
//   out[n,:]  = sum_m exp(s[n,m]-colmax[m]) * (1/colsum[m]) * V[m,:]
// Two-pass flash-style recompute:
//   - bf16 WMMA (v_wmma_f32_16x16x32_bf16), f32 softmax math
//   - 8 waves/WG share streamed tiles staged in LDS via
//     global_load_async_to_lds_b128 (ASYNCcnt), DOUBLE-BUFFERED so the
//     async DMA of chunk i+1 overlaps the WMMAs of chunk i
//   - pass 2 computes S^T = K·Q^T so P^T feeds the PV WMMA B operand in-register

#define NQ 8192
#define NK 8192
#define DD 128
#define G1 8    // pass-1 split over N
#define G2 16   // pass-2 split over M

typedef __attribute__((ext_vector_type(16))) __bf16 v16bf;
typedef __attribute__((ext_vector_type(8)))  float  v8f;

union BF16Frag { v16bf v; unsigned int u[8]; };

__device__ __forceinline__ unsigned short f32_to_bf16_bits(float f) {
  unsigned int u = __float_as_uint(f);
  u += 0x7FFFu + ((u >> 16) & 1u);   // round-to-nearest-even
  return (unsigned short)(u >> 16);
}

// Load one 16-bit WMMA operand fragment (A-style layout, B assumed symmetric):
// lane half hi: elements 0..7 = row[k0 + 8*hi .. +7], 8..15 = row[k0+16+8*hi .. +7]
__device__ __forceinline__ v16bf load_frag(const unsigned short* rowp, int k0, int hi) {
  BF16Frag r;
  uint4 x = *reinterpret_cast<const uint4*>(rowp + k0 + 8 * hi);
  uint4 y = *reinterpret_cast<const uint4*>(rowp + k0 + 16 + 8 * hi);
  r.u[0] = x.x; r.u[1] = x.y; r.u[2] = x.z; r.u[3] = x.w;
  r.u[4] = y.x; r.u[5] = y.y; r.u[6] = y.z; r.u[7] = y.w;
  return r.v;
}

// Async copy of one 16-byte segment: global -> LDS, tracked by ASYNCcnt.
__device__ __forceinline__ void async_ld16(unsigned short* lds_dst,
                                           const unsigned short* gsrc) {
  unsigned ldsoff = (unsigned)(unsigned long long)lds_dst;  // low 32 bits = LDS offset
  asm volatile("global_load_async_to_lds_b128 %0, %1, off"
               :: "v"(ldsoff), "v"(gsrc) : "memory");
}
__device__ __forceinline__ void wait_async0() {
  asm volatile("s_wait_asynccnt 0x0" ::: "memory");
}

// ---- Kernel 0: convert Q,K -> bf16 row-major; V -> bf16 transposed; zero out ----
__global__ void attn_prep(const float* __restrict__ Q, const float* __restrict__ K,
                          const float* __restrict__ V,
                          unsigned short* __restrict__ Qb, unsigned short* __restrict__ Kb,
                          unsigned short* __restrict__ Vt, float* __restrict__ out) {
  int i = blockIdx.x * 256 + threadIdx.x;          // i < NQ*DD (== NK*DD)
  Qb[i] = f32_to_bf16_bits(Q[i]);
  Kb[i] = f32_to_bf16_bits(K[i]);
  int m = i >> 7, d = i & 127;
  Vt[d * NK + m] = f32_to_bf16_bits(V[i]);         // Vt[d][m] = V[m][d]
  out[i] = 0.f;
}

// ---- Kernel 1: per-key-column partial (max, sum) over an N-slice ----
// 8 waves/WG, wave w owns key block m0 = blk*128 + w*16; Q tiles double-buffered in LDS.
__global__ void __launch_bounds__(256)
attn_pass1(const unsigned short* __restrict__ Qb,
           const unsigned short* __restrict__ Kb,
           float* __restrict__ pmax, float* __restrict__ psum) {
  __shared__ unsigned short sQ[2][32 * DD];            // 2 x 8 KB ping-pong
  const int tid  = threadIdx.x;
  const int lane = tid & 31, w = tid >> 5;
  const int lm = lane & 15, hi = lane >> 4;
  const int m0 = blockIdx.x * 128 + w * 16;
  const int g  = blockIdx.y;
  const int e0 = tid * 2;                              // this thread's segments

  v16bf kf[4];                                         // B: column m = lane%16
  const unsigned short* krow = Kb + (m0 + lm) * DD;
#pragma unroll
  for (int kc = 0; kc < 4; ++kc) kf[kc] = load_frag(krow, kc * 32, hi);

  float rm = -INFINITY, rs = 0.f;
  const int nBeg = g * (NQ / G1);
  const int nIter = (NQ / G1) / 32;

  // prologue: stage chunk 0
#pragma unroll
  for (int j = 0; j < 2; ++j) {
    int e = e0 + j;
    async_ld16(sQ[0] + e * 8, Qb + (nBeg + (e >> 4)) * DD + (e & 15) * 8);
  }
  wait_async0();
  __syncthreads();

  for (int it = 0; it < nIter; ++it) {
    const int cur = it & 1;
    if (it + 1 < nIter) {                              // stage next chunk (overlapped)
      const int n1 = nBeg + (it + 1) * 32;
#pragma unroll
      for (int j = 0; j < 2; ++j) {
        int e = e0 + j;
        async_ld16(sQ[1 - cur] + e * 8, Qb + (n1 + (e >> 4)) * DD + (e & 15) * 8);
      }
    }
    // batch all fragment loads, then the WMMA chains
    v16bf qa0[4], qa1[4];
    const unsigned short* q0 = sQ[cur] + lm * DD;
    const unsigned short* q1 = sQ[cur] + (16 + lm) * DD;
#pragma unroll
    for (int kc = 0; kc < 4; ++kc) qa0[kc] = load_frag(q0, kc * 32, hi);
#pragma unroll
    for (int kc = 0; kc < 4; ++kc) qa1[kc] = load_frag(q1, kc * 32, hi);

#pragma unroll
    for (int t = 0; t < 2; ++t) {                      // two 16-row S tiles
      v8f c;
#pragma unroll
      for (int r = 0; r < 8; ++r) c[r] = 0.f;
#pragma unroll
      for (int kc = 0; kc < 4; ++kc)
        c = __builtin_amdgcn_wmma_f32_16x16x32_bf16(false, t ? qa1[kc] : qa0[kc],
                                                    false, kf[kc],
                                                    (short)0, c, false, false);
      float tm = c[0];
#pragma unroll
      for (int r = 1; r < 8; ++r) tm = fmaxf(tm, c[r]);
      float nm = fmaxf(rm, tm);
      float s = 0.f;
#pragma unroll
      for (int r = 0; r < 8; ++r) s += __expf(c[r] - nm);
      rs = rs * __expf(rm - nm) + s;
      rm = nm;
    }
    wait_async0();                                     // next buffer landed
    __syncthreads();                                   // everyone done with cur
  }
  // combine the two lane halves (rows 0-7 vs 8-15 of each tile)
  float om = __shfl_xor(rm, 16, 32);
  float os = __shfl_xor(rs, 16, 32);
  float nm = fmaxf(rm, om);
  float ns = rs * __expf(rm - nm) + os * __expf(om - nm);
  if (lane < 16) {
    pmax[g * NK + m0 + lm] = nm;
    psum[g * NK + m0 + lm] = ns;
  }
}

// ---- Kernel 2: reduce G1 partials per column with rescale ----
__global__ void attn_pass1b(const float* __restrict__ pmax, const float* __restrict__ psum,
                            float* __restrict__ colmax, float* __restrict__ rcolsum) {
  int m = blockIdx.x * 256 + threadIdx.x;
  float bm = -INFINITY;
#pragma unroll
  for (int g = 0; g < G1; ++g) bm = fmaxf(bm, pmax[g * NK + m]);
  float s = 0.f;
#pragma unroll
  for (int g = 0; g < G1; ++g) s += psum[g * NK + m] * __expf(pmax[g * NK + m] - bm);
  colmax[m]  = bm;
  rcolsum[m] = 1.0f / s;
}

// ---- Kernel 3: out += P·V. 8 waves/WG, wave w owns queries n0 = blk*128 + w*16;
//      K and V^T chunks double-buffered in LDS, shared by all 8 waves. ----
__global__ void __launch_bounds__(256)
attn_pass2(const unsigned short* __restrict__ Qb,
           const unsigned short* __restrict__ Kb,
           const unsigned short* __restrict__ Vt,
           const float* __restrict__ colmax,
           const float* __restrict__ rcolsum,
           float* __restrict__ out) {
  __shared__ unsigned short sK[2][32 * DD];            // 2 x 8 KB
  __shared__ unsigned short sV[2][DD * 32];            // 2 x 8 KB
  const int tid  = threadIdx.x;
  const int lane = tid & 31, w = tid >> 5;
  const int lm = lane & 15, hi = lane >> 4;
  const int n0 = blockIdx.x * 128 + w * 16;
  const int mBeg = blockIdx.y * (NK / G2);
  const int mIter = (NK / G2) / 32;
  const int e0 = tid * 2;

  v16bf qf[4];                                         // B of S^T: column n = lane%16
  const unsigned short* qrow = Qb + (n0 + lm) * DD;
#pragma unroll
  for (int kc = 0; kc < 4; ++kc) qf[kc] = load_frag(qrow, kc * 32, hi);

  v8f acc[8];                                          // out^T accum: 8 d-tiles
#pragma unroll
  for (int dt = 0; dt < 8; ++dt)
#pragma unroll
    for (int r = 0; r < 8; ++r) acc[dt][r] = 0.f;

  // prologue: stage chunk 0
#pragma unroll
  for (int j = 0; j < 2; ++j) {
    int e = e0 + j;
    async_ld16(sK[0] + e * 8, Kb + (mBeg + (e >> 4)) * DD + (e & 15) * 8);
    async_ld16(sV[0] + e * 8, Vt + (e >> 2) * NK + mBeg + (e & 3) * 8);
  }
  wait_async0();
  __syncthreads();

  for (int it = 0; it < mIter; ++it) {
    const int cur = it & 1;
    const int mc = mBeg + it * 32;
    if (it + 1 < mIter) {                              // stage next chunk (overlapped)
      const int m1 = mc + 32;
#pragma unroll
      for (int j = 0; j < 2; ++j) {
        int e = e0 + j;
        async_ld16(sK[1 - cur] + e * 8, Kb + (m1 + (e >> 4)) * DD + (e & 15) * 8);
        async_ld16(sV[1 - cur] + e * 8, Vt + (e >> 2) * NK + m1 + (e & 3) * 8);
      }
    }
    // batch all K fragment loads, then the S^T WMMA chains
    v16bf ka1[4], ka2[4];
    const unsigned short* krow1 = sK[cur] + lm * DD;         // A rows m (local 0..15)
    const unsigned short* krow2 = sK[cur] + (16 + lm) * DD;  // A rows m (local 16..31)
#pragma unroll
    for (int kc = 0; kc < 4; ++kc) ka1[kc] = load_frag(krow1, kc * 32, hi);
#pragma unroll
    for (int kc = 0; kc < 4; ++kc) ka2[kc] = load_frag(krow2, kc * 32, hi);

    v8f s1, s2;
#pragma unroll
    for (int r = 0; r < 8; ++r) { s1[r] = 0.f; s2[r] = 0.f; }
#pragma unroll
    for (int kc = 0; kc < 4; ++kc)
      s1 = __builtin_amdgcn_wmma_f32_16x16x32_bf16(false, ka1[kc], false, qf[kc],
                                                   (short)0, s1, false, false);
#pragma unroll
    for (int kc = 0; kc < 4; ++kc)
      s2 = __builtin_amdgcn_wmma_f32_16x16x32_bf16(false, ka2[kc], false, qf[kc],
                                                   (short)0, s2, false, false);

    // s1[r] = S^T[mc + 8*hi + r, n0+lm],  s2[r] = S^T[mc+16+8*hi + r, n0+lm]
    const float* cm1 = colmax  + mc + 8 * hi;
    const float* rs1 = rcolsum + mc + 8 * hi;
    const float* cm2 = colmax  + mc + 16 + 8 * hi;
    const float* rs2 = rcolsum + mc + 16 + 8 * hi;
    BF16Frag pb;                                       // P^T packed straight into B layout
#pragma unroll
    for (int j = 0; j < 4; ++j) {
      float a0 = __expf(s1[2 * j]     - cm1[2 * j])     * rs1[2 * j];
      float a1 = __expf(s1[2 * j + 1] - cm1[2 * j + 1]) * rs1[2 * j + 1];
      pb.u[j] = (unsigned int)f32_to_bf16_bits(a0) |
                ((unsigned int)f32_to_bf16_bits(a1) << 16);
      float b0 = __expf(s2[2 * j]     - cm2[2 * j])     * rs2[2 * j];
      float b1 = __expf(s2[2 * j + 1] - cm2[2 * j + 1]) * rs2[2 * j + 1];
      pb.u[4 + j] = (unsigned int)f32_to_bf16_bits(b0) |
                    ((unsigned int)f32_to_bf16_bits(b1) << 16);
    }
    // batch all V^T fragment loads, then the PV WMMA chain
    v16bf va[8];
#pragma unroll
    for (int dt = 0; dt < 8; ++dt)
      va[dt] = load_frag(sV[cur] + (dt * 16 + lm) * 32, 0, hi);
    // out^T(16d x 16n) += V^T(16d x 32m) @ P^T(32m x 16n)
#pragma unroll
    for (int dt = 0; dt < 8; ++dt)
      acc[dt] = __builtin_amdgcn_wmma_f32_16x16x32_bf16(false, va[dt], false, pb.v,
                                                        (short)0, acc[dt], false, false);
    wait_async0();                                     // next buffer landed
    __syncthreads();                                   // everyone done with cur
  }
  // acc[dt][r] = out[n0+lm, dt*16 + 8*hi + r]; combine the G2 m-slices atomically
  float* obase = out + (n0 + lm) * DD + 8 * hi;
#pragma unroll
  for (int dt = 0; dt < 8; ++dt)
#pragma unroll
    for (int r = 0; r < 8; ++r)
      unsafeAtomicAdd(obase + dt * 16 + r, acc[dt][r]);
}

extern "C" void kernel_launch(void* const* d_in, const int* in_sizes, int n_in,
                              void* d_out, int out_size, void* d_ws, size_t ws_size,
                              hipStream_t stream) {
  (void)in_sizes; (void)n_in; (void)out_size; (void)ws_size;
  const float* Q = (const float*)d_in[0];
  const float* K = (const float*)d_in[1];
  const float* V = (const float*)d_in[2];
  float* out = (float*)d_out;

  unsigned short* Qb = (unsigned short*)d_ws;        // 2 MB
  unsigned short* Kb = Qb + NQ * DD;                 // 2 MB
  unsigned short* Vt = Kb + NK * DD;                 // 2 MB (transposed V)
  float* pmax    = (float*)(Vt + NK * DD);           // G1*NK
  float* psum    = pmax + G1 * NK;                   // G1*NK
  float* colmax  = psum + G1 * NK;                   // NK
  float* rcolsum = colmax + NK;                      // NK

  attn_prep<<<(NQ * DD) / 256, 256, 0, stream>>>(Q, K, V, Qb, Kb, Vt, out);
  attn_pass1<<<dim3(NK / 128, G1), 256, 0, stream>>>(Qb, Kb, pmax, psum);
  attn_pass1b<<<NK / 256, 256, 0, stream>>>(pmax, psum, colmax, rcolsum);
  attn_pass2<<<dim3(NQ / 128, G2), 256, 0, stream>>>(Qb, Kb, Vt, colmax, rcolsum, out);
}
